// EmbedLayerUpdate_927712936014
// MI455X (gfx1250) — matrix operand order repaired and checked by reading, not compiled, not run on hardware
//
#include <hip/hip_runtime.h>

// ---------------------------------------------------------------------------
// Types for CDNA5 WMMA (gfx1250, wave32)
// ---------------------------------------------------------------------------
typedef __attribute__((ext_vector_type(16))) __bf16 v16bf;
typedef __attribute__((ext_vector_type(8)))  __bf16 v8bf;
typedef __attribute__((ext_vector_type(2)))  __bf16 v2bf;
typedef __attribute__((ext_vector_type(8)))  float  v8f;
typedef __attribute__((ext_vector_type(4)))  unsigned int u32x4;

// native fp32 -> bf16 (hardware cvt, RNE)
__device__ __forceinline__ unsigned short bf(float x) {
  return __builtin_bit_cast(unsigned short, (__bf16)x);
}
// two fp32 -> packed bf16x2 dword (pattern-matches v_cvt_pk_bf16_f32)
__device__ __forceinline__ unsigned pk2(float lo, float hi) {
  v2bf p;
  p[0] = (__bf16)lo;
  p[1] = (__bf16)hi;
  return __builtin_bit_cast(unsigned, p);
}
__device__ __forceinline__ float f4get(const float4& f, int e) {
  return e == 0 ? f.x : e == 1 ? f.y : e == 2 ? f.z : f.w;
}
__device__ __forceinline__ v8bf cvt8(float4 a, float4 b) {
  v8bf r;
  r[0] = (__bf16)a.x; r[1] = (__bf16)a.y; r[2] = (__bf16)a.z; r[3] = (__bf16)a.w;
  r[4] = (__bf16)b.x; r[5] = (__bf16)b.y; r[6] = (__bf16)b.z; r[7] = (__bf16)b.w;
  return r;
}

// A fragment: 16x32 bf16 tile, row-major [m][k] in LDS (pitch mult of 8 shorts).
// ISA: lanes 0-15 -> K=0..7 (v0-3), K=16..23 (v4-7); lanes 16-31 -> +8.
__device__ __forceinline__ v16bf frag_a(const unsigned short* tile, int pitch) {
  const int l = threadIdx.x & 31;
  const unsigned short* p = tile + (l & 15) * pitch + ((l & 16) ? 8 : 0);
  const u32x4* q = (const u32x4*)p;
  u32x4 d0 = q[0], d1 = q[2];          // +16 shorts => K advance of 16
  v8bf lo = __builtin_bit_cast(v8bf, d0);
  v8bf hi = __builtin_bit_cast(v8bf, d1);
  v16bf r;
#pragma unroll
  for (int e = 0; e < 8; ++e) { r[e] = lo[e]; r[e + 8] = hi[e]; }
  return r;
}

// B fragment: 32x16 tile stored TRANSPOSED in LDS as [n][k].
__device__ __forceinline__ v16bf frag_b(const unsigned short* tileT, int pitch) {
  const int l = threadIdx.x & 31;
  const unsigned short* p = tileT + (l & 15) * pitch + ((l & 16) ? 8 : 0);
  const u32x4* q = (const u32x4*)p;
  u32x4 d0 = q[0], d1 = q[2];
  v8bf lo = __builtin_bit_cast(v8bf, d0);
  v8bf hi = __builtin_bit_cast(v8bf, d1);
  v16bf r;
#pragma unroll
  for (int e = 0; e < 8; ++e) { r[e] = lo[e]; r[e + 8] = hi[e]; }
  return r;
}

__device__ __forceinline__ v8f wmma_bf16(v16bf a, v16bf b, v8f c) {
  return __builtin_amdgcn_wmma_f32_16x16x32_bf16(false, a, false, b,
                                                 (short)0, c, false, false);
}

// ---------------------------------------------------------------------------
// Forward GEMM: out[b,m,n] = sum_k W[m,k] * X[b,k,n] + bias[m]
//   W: [M,K], X: [B,K,64], out: [B,M,64].  Tile 128x64, K-step 64, 8 waves.
// ---------------------------------------------------------------------------
__global__ __launch_bounds__(256) void k_gemm_fwd(
    const float* __restrict__ W, const float* __restrict__ bias,
    const float* __restrict__ X, float* __restrict__ out, int M, int K) {
  __shared__ __align__(16) unsigned short lA[128 * 72];  // [m][k], pitch 72
  __shared__ __align__(16) unsigned short lB[64 * 72];   // [n][k] transposed

  const int b = blockIdx.y;
  const int mBase = blockIdx.x * 128;
  const int t = threadIdx.x;
  const int rowBase = (t >> 5) * 16;
  const float* Xb = X + (size_t)b * K * 64;

  const int ar = t >> 1, acs = (t & 1) * 32;       // A: 128 rows x 64 k
  const int bk = (t >> 3) * 2, bns = (t & 7) * 8;  // B: k-pairs x 8 n

  v8f acc[4] = {};

  for (int k0 = 0; k0 < K; k0 += 64) {
    {  // stage A (contiguous -> b128 LDS stores)
      const float4* s4 = (const float4*)(W + (size_t)(mBase + ar) * K + k0 + acs);
      unsigned short* dst = lA + ar * 72 + acs;
#pragma unroll
      for (int q = 0; q < 4; ++q)
        *(v8bf*)(dst + q * 8) = cvt8(s4[2 * q], s4[2 * q + 1]);
      if (k0 + 64 < K)
        __builtin_prefetch(W + (size_t)(mBase + ar) * K + k0 + 64 + acs, 0, 1);
    }
    {  // stage B transposed: two k-rows per thread, packed b32 stores
      const float4* s0 = (const float4*)(Xb + (size_t)(k0 + bk) * 64 + bns);
      const float4* s1 = (const float4*)(Xb + (size_t)(k0 + bk + 1) * 64 + bns);
      float4 a0 = s0[0], a1 = s0[1], c0 = s1[0], c1 = s1[1];
#pragma unroll
      for (int j = 0; j < 8; ++j) {
        float v0 = j < 4 ? f4get(a0, j & 3) : f4get(a1, j & 3);
        float v1 = j < 4 ? f4get(c0, j & 3) : f4get(c1, j & 3);
        *(unsigned*)(lB + (bns + j) * 72 + bk) = pk2(v0, v1);
      }
    }
    __syncthreads();
#pragma unroll
    for (int kk = 0; kk < 64; kk += 32) {
      v16bf a = frag_a(lA + rowBase * 72 + kk, 72);
#pragma unroll
      for (int n = 0; n < 4; ++n)
        acc[n] = wmma_bf16(a, frag_b(lB + n * 16 * 72 + kk, 72), acc[n]);
    }
    __syncthreads();
  }

  const int l = t & 31;
  const int col = l & 15;
  const int r0 = rowBase + ((l & 16) ? 8 : 0);
  float bv[8];
#pragma unroll
  for (int v = 0; v < 8; ++v) bv[v] = bias[mBase + r0 + v];
#pragma unroll
  for (int n = 0; n < 4; ++n)
#pragma unroll
    for (int v = 0; v < 8; ++v)
      out[((size_t)b * M + (mBase + r0 + v)) * 64 + n * 16 + col] = acc[n][v] + bv[v];
}

// ---------------------------------------------------------------------------
// Backward GEMM: out[b,c,n] = sum_k (X[b,k,n] - bias[k]) * W[k,c]
//   W: [K,M] (row stride M), X: [B,K,64], out: [B,M,64]
// ---------------------------------------------------------------------------
__global__ __launch_bounds__(256) void k_gemm_bwd(
    const float* __restrict__ W, const float* __restrict__ bias,
    const float* __restrict__ X, float* __restrict__ out, int M, int K) {
  __shared__ __align__(16) unsigned short lA[128 * 72];  // [c][k] = W[k, mBase+c]
  __shared__ __align__(16) unsigned short lB[64 * 72];   // [n][k]

  const int b = blockIdx.y;
  const int mBase = blockIdx.x * 128;
  const int t = threadIdx.x;
  const int rowBase = (t >> 5) * 16;
  const float* Xb = X + (size_t)b * K * 64;

  const int ak = (t & 31) * 2, acs = (t >> 5) * 16;  // A: 32 k-pairs x 16 c
  const int bk = (t >> 3) * 2, bns = (t & 7) * 8;    // B: k-pairs x 8 n

  v8f acc[4] = {};

  for (int k0 = 0; k0 < K; k0 += 64) {
    {  // stage A transposed: rows of W contiguous in c, pack k-pairs -> b32
      const float4* p0 = (const float4*)(W + (size_t)(k0 + ak) * M + mBase + acs);
      const float4* p1 = (const float4*)(W + (size_t)(k0 + ak + 1) * M + mBase + acs);
      float4 f0[4] = {p0[0], p0[1], p0[2], p0[3]};
      float4 f1[4] = {p1[0], p1[1], p1[2], p1[3]};
#pragma unroll
      for (int c = 0; c < 16; ++c)
        *(unsigned*)(lA + (acs + c) * 72 + ak) =
            pk2(f4get(f0[c >> 2], c & 3), f4get(f1[c >> 2], c & 3));
      if (k0 + 64 < K)
        __builtin_prefetch(W + (size_t)(k0 + 64 + ak) * M + mBase + acs, 0, 1);
    }
    {  // stage B with bias subtraction
      const float bk0 = bias[k0 + bk], bk1 = bias[k0 + bk + 1];
      const float4* s0 = (const float4*)(Xb + (size_t)(k0 + bk) * 64 + bns);
      const float4* s1 = (const float4*)(Xb + (size_t)(k0 + bk + 1) * 64 + bns);
      float4 a0 = s0[0], a1 = s0[1], c0 = s1[0], c1 = s1[1];
#pragma unroll
      for (int j = 0; j < 8; ++j) {
        float v0 = (j < 4 ? f4get(a0, j & 3) : f4get(a1, j & 3)) - bk0;
        float v1 = (j < 4 ? f4get(c0, j & 3) : f4get(c1, j & 3)) - bk1;
        *(unsigned*)(lB + (bns + j) * 72 + bk) = pk2(v0, v1);
      }
    }
    __syncthreads();
#pragma unroll
    for (int kk = 0; kk < 64; kk += 32) {
      v16bf a = frag_a(lA + rowBase * 72 + kk, 72);
#pragma unroll
      for (int n = 0; n < 4; ++n)
        acc[n] = wmma_bf16(a, frag_b(lB + n * 16 * 72 + kk, 72), acc[n]);
    }
    __syncthreads();
  }

  const int l = t & 31;
  const int col = l & 15;
  const int r0 = rowBase + ((l & 16) ? 8 : 0);
#pragma unroll
  for (int n = 0; n < 4; ++n)
#pragma unroll
    for (int v = 0; v < 8; ++v)
      out[((size_t)b * M + (mBase + r0 + v)) * 64 + n * 16 + col] = acc[n][v];
}

// ---------------------------------------------------------------------------
// Forward row-MLP (65536 rows): info(3)->fc1 relu->fc1_1 ; [pre|emb]->fc2 relu
// ->fc2_1 -> outMu.  128 threads (4 waves), 16 rows/wave.
// ---------------------------------------------------------------------------
__global__ __launch_bounds__(128) void k_mlp_fwd(
    const float* __restrict__ pre, const float* __restrict__ lb,
    const float* __restrict__ ub,
    const float* __restrict__ fc1_w,  const float* __restrict__ fc1_b,
    const float* __restrict__ fc11_w, const float* __restrict__ fc11_b,
    const float* __restrict__ fc2_w,  const float* __restrict__ fc2_b,
    const float* __restrict__ fc21_w, const float* __restrict__ fc21_b,
    float* __restrict__ outMu) {
  __shared__ __align__(16) unsigned short w11[64 * 64];
  __shared__ __align__(16) unsigned short w2[64 * 128];
  __shared__ __align__(16) unsigned short w21[64 * 64];
  __shared__ __align__(16) unsigned short stage[4][16 * 128];

  const int t = threadIdx.x;
  for (int i = t; i < 64 * 64;  i += 128) w11[i] = bf(fc11_w[i]);
  for (int i = t; i < 64 * 128; i += 128) w2[i]  = bf(fc2_w[i]);
  for (int i = t; i < 64 * 64;  i += 128) w21[i] = bf(fc21_w[i]);

  const int wave = t >> 5, l = t & 31;
  const int m = l & 15;
  const int jb = (l & 16) ? 32 : 0;
  const int rBase = blockIdx.x * 64 + wave * 16;
  const int r = rBase + m;
  unsigned short* st = &stage[wave][0];      // pitch 128
  const int col0 = l & 15;
  const int rr0 = (l & 16) ? 8 : 0;

  // phase 0: ratio features + fc1 (K=3) + relu -> st[:,0:64]
  {
    const float lbv = lb[r], ubv = ub[r];
    const float lt = fminf(lbv, 0.f);
    const float ut = fmaxf(ubv, 0.f);
    const float diff = ut - lt;
    const float urr = ut / (diff + 1e-8f);
    const float lrr = 1.f - urr;
#pragma unroll
    for (int g = 0; g < 4; ++g) {
      v8bf pk;
#pragma unroll
      for (int e = 0; e < 8; ++e) {
        const int jj = jb + g * 8 + e;
        float h = fc1_w[jj * 3] * urr + fc1_w[jj * 3 + 1] * lrr +
                  fc1_w[jj * 3 + 2] * diff + fc1_b[jj];
        pk[e] = (__bf16)fmaxf(h, 0.f);
      }
      *(v8bf*)(st + m * 128 + jb + g * 8) = pk;
    }
  }
  __syncthreads();

  // phase 1: emb = h1 @ fc1_1^T + b  (K=64)
  v8f acc[4] = {};
#pragma unroll
  for (int kk = 0; kk < 64; kk += 32) {
    v16bf a = frag_a(st + kk, 128);
#pragma unroll
    for (int n = 0; n < 4; ++n)
      acc[n] = wmma_bf16(a, frag_b(w11 + n * 16 * 64 + kk, 64), acc[n]);
  }
  __syncthreads();
#pragma unroll
  for (int n = 0; n < 4; ++n) {
    const int j = n * 16 + col0;
    const float bj = fc11_b[j];
#pragma unroll
    for (int v = 0; v < 8; ++v)
      st[(rr0 + v) * 128 + 64 + j] = bf(acc[n][v] + bj);
  }
  {  // load pre row -> st[:,0:64]
    const float4* src = (const float4*)(pre + (size_t)r * 64 + jb);
#pragma unroll
    for (int q = 0; q < 4; ++q)
      *(v8bf*)(st + m * 128 + jb + q * 8) = cvt8(src[2 * q], src[2 * q + 1]);
  }
  __syncthreads();

  // phase 2: h2 = relu([pre|emb] @ fc2^T + b)  (K=128)
#pragma unroll
  for (int n = 0; n < 4; ++n) acc[n] = (v8f){};
#pragma unroll
  for (int kk = 0; kk < 128; kk += 32) {
    v16bf a = frag_a(st + kk, 128);
#pragma unroll
    for (int n = 0; n < 4; ++n)
      acc[n] = wmma_bf16(a, frag_b(w2 + n * 16 * 128 + kk, 128), acc[n]);
  }
  __syncthreads();
#pragma unroll
  for (int n = 0; n < 4; ++n) {
    const int j = n * 16 + col0;
    const float bj = fc2_b[j];
#pragma unroll
    for (int v = 0; v < 8; ++v)
      st[(rr0 + v) * 128 + j] = bf(fmaxf(acc[n][v] + bj, 0.f));
  }
  __syncthreads();

  // phase 3: out = h2 @ fc2_1^T + b  (K=64)
#pragma unroll
  for (int n = 0; n < 4; ++n) acc[n] = (v8f){};
#pragma unroll
  for (int kk = 0; kk < 64; kk += 32) {
    v16bf a = frag_a(st + kk, 128);
#pragma unroll
    for (int n = 0; n < 4; ++n)
      acc[n] = wmma_bf16(a, frag_b(w21 + n * 16 * 64 + kk, 64), acc[n]);
  }
#pragma unroll
  for (int n = 0; n < 4; ++n) {
    const int j = n * 16 + col0;
    const float bj = fc21_b[j];
#pragma unroll
    for (int v = 0; v < 8; ++v)
      outMu[(size_t)(rBase + rr0 + v) * 64 + j] = acc[n][v] + bj;
  }
}

// ---------------------------------------------------------------------------
// Backward row-MLP: x=[preT|mu] -> fc3 relu -> fc3_1 -> mu (in place)
// ---------------------------------------------------------------------------
__global__ __launch_bounds__(128) void k_mlp_bwd(
    const float* __restrict__ preT, float* mu,
    const float* __restrict__ fc3_w,  const float* __restrict__ fc3_b,
    const float* __restrict__ fc31_w, const float* __restrict__ fc31_b) {
  __shared__ __align__(16) unsigned short w3[64 * 128];
  __shared__ __align__(16) unsigned short w31[64 * 64];
  __shared__ __align__(16) unsigned short stage[4][16 * 128];

  const int t = threadIdx.x;
  for (int i = t; i < 64 * 128; i += 128) w3[i]  = bf(fc3_w[i]);
  for (int i = t; i < 64 * 64;  i += 128) w31[i] = bf(fc31_w[i]);

  const int wave = t >> 5, l = t & 31;
  const int m = l & 15;
  const int jb = (l & 16) ? 32 : 0;
  const int rBase = blockIdx.x * 64 + wave * 16;
  const int r = rBase + m;
  unsigned short* st = &stage[wave][0];
  const int col0 = l & 15;
  const int rr0 = (l & 16) ? 8 : 0;

  {  // stage x = [preT | mu]
    const float4* s0 = (const float4*)(preT + (size_t)r * 64 + jb);
    const float4* s1 = (const float4*)(mu + (size_t)r * 64 + jb);
#pragma unroll
    for (int q = 0; q < 4; ++q)
      *(v8bf*)(st + m * 128 + jb + q * 8) = cvt8(s0[2 * q], s0[2 * q + 1]);
#pragma unroll
    for (int q = 0; q < 4; ++q)
      *(v8bf*)(st + m * 128 + 64 + jb + q * 8) = cvt8(s1[2 * q], s1[2 * q + 1]);
  }
  __syncthreads();

  // fc3: K=128, relu
  v8f acc[4] = {};
#pragma unroll
  for (int kk = 0; kk < 128; kk += 32) {
    v16bf a = frag_a(st + kk, 128);
#pragma unroll
    for (int n = 0; n < 4; ++n)
      acc[n] = wmma_bf16(a, frag_b(w3 + n * 16 * 128 + kk, 128), acc[n]);
  }
  __syncthreads();
#pragma unroll
  for (int n = 0; n < 4; ++n) {
    const int j = n * 16 + col0;
    const float bj = fc3_b[j];
#pragma unroll
    for (int v = 0; v < 8; ++v)
      st[(rr0 + v) * 128 + j] = bf(fmaxf(acc[n][v] + bj, 0.f));
  }
  __syncthreads();

  // fc3_1: K=64 -> mu
#pragma unroll
  for (int n = 0; n < 4; ++n) acc[n] = (v8f){};
#pragma unroll
  for (int kk = 0; kk < 64; kk += 32) {
    v16bf a = frag_a(st + kk, 128);
#pragma unroll
    for (int n = 0; n < 4; ++n)
      acc[n] = wmma_bf16(a, frag_b(w31 + n * 16 * 64 + kk, 64), acc[n]);
  }
#pragma unroll
  for (int n = 0; n < 4; ++n) {
    const int j = n * 16 + col0;
    const float bj = fc31_b[j];
#pragma unroll
    for (int v = 0; v < 8; ++v)
      mu[(size_t)(rBase + rr0 + v) * 64 + j] = acc[n][v] + bj;
  }
}

// ---------------------------------------------------------------------------
extern "C" void kernel_launch(void* const* d_in, const int* in_sizes, int n_in,
                              void* d_out, int out_size, void* d_ws, size_t ws_size,
                              hipStream_t stream) {
  (void)in_sizes; (void)n_in; (void)out_size; (void)ws_size;
  const float* mu0 = (const float*)d_in[0];
  const float* lb1 = (const float*)d_in[6];
  const float* ub1 = (const float*)d_in[7];
  const float* lb2 = (const float*)d_in[8];
  const float* ub2 = (const float*)d_in[9];
  const float* lb3 = (const float*)d_in[10];
  const float* ub3 = (const float*)d_in[11];
  const float* W1 = (const float*)d_in[12]; const float* b1 = (const float*)d_in[13];
  const float* W2 = (const float*)d_in[14]; const float* b2 = (const float*)d_in[15];
  const float* W3 = (const float*)d_in[16]; const float* b3 = (const float*)d_in[17];
  const float* fc1_w  = (const float*)d_in[18]; const float* fc1_b  = (const float*)d_in[19];
  const float* fc11_w = (const float*)d_in[20]; const float* fc11_b = (const float*)d_in[21];
  const float* fc2_w  = (const float*)d_in[22]; const float* fc2_b  = (const float*)d_in[23];
  const float* fc21_w = (const float*)d_in[24]; const float* fc21_b = (const float*)d_in[25];
  const float* fc3_w  = (const float*)d_in[26]; const float* fc3_b  = (const float*)d_in[27];
  const float* fc31_w = (const float*)d_in[28]; const float* fc31_b = (const float*)d_in[29];

  float* out = (float*)d_out;
  float* outMu0 = out;                 // [32,1024,64]
  float* outMu1 = out + 2097152;       // [32,2048,64]
  float* outMu2 = outMu1 + 4194304;
  float* outMu3 = outMu2 + 4194304;
  float* pre = (float*)d_ws;           // 32*2048*64 fp32 = 16.8 MB scratch

  hipMemcpyAsync(outMu0, mu0, (size_t)2097152 * sizeof(float),
                 hipMemcpyDeviceToDevice, stream);

  dim3 gG(16, 32), bG(256);
  dim3 gM(1024),  bM(128);

  // forward sweep
  k_gemm_fwd<<<gG, bG, 0, stream>>>(W1, b1, mu0, pre, 2048, 1024);
  k_mlp_fwd<<<gM, bM, 0, stream>>>(pre, lb1, ub1, fc1_w, fc1_b, fc11_w, fc11_b,
                                   fc2_w, fc2_b, fc21_w, fc21_b, outMu1);
  k_gemm_fwd<<<gG, bG, 0, stream>>>(W2, b2, outMu1, pre, 2048, 2048);
  k_mlp_fwd<<<gM, bM, 0, stream>>>(pre, lb2, ub2, fc1_w, fc1_b, fc11_w, fc11_b,
                                   fc2_w, fc2_b, fc21_w, fc21_b, outMu2);
  k_gemm_fwd<<<gG, bG, 0, stream>>>(W3, b3, outMu2, pre, 2048, 2048);
  k_mlp_fwd<<<gM, bM, 0, stream>>>(pre, lb3, ub3, fc1_w, fc1_b, fc11_w, fc11_b,
                                   fc2_w, fc2_b, fc21_w, fc21_b, outMu3);

  // backward sweep
  k_gemm_bwd<<<gG, bG, 0, stream>>>(W3, b3, outMu3, pre, 2048, 2048);
  k_mlp_bwd<<<gM, bM, 0, stream>>>(pre, outMu2, fc3_w, fc3_b, fc31_w, fc31_b);
  k_gemm_bwd<<<gG, bG, 0, stream>>>(W2, b2, outMu2, pre, 2048, 2048);
  k_mlp_bwd<<<gM, bM, 0, stream>>>(pre, outMu1, fc3_w, fc3_b, fc31_w, fc31_b);
}
